// GateDeepSeekV3_5282809775020
// MI455X (gfx1250) — matrix operand order repaired and checked by reading, not compiled, not run on hardware
//
#include <hip/hip_runtime.h>
#include <hip/hip_bf16.h>
#include <math.h>

typedef __attribute__((ext_vector_type(16))) __bf16 v16bf;
typedef __attribute__((ext_vector_type(8)))  __bf16 v8bf;
typedef __attribute__((ext_vector_type(4)))  __bf16 v4bf;
typedef __attribute__((ext_vector_type(8)))  float  v8f;
typedef __attribute__((ext_vector_type(4)))  float  v4f;

#define TOKENS   8192
#define DIM      7168
#define EXPERTS  256
#define KCHUNK   32
#define NITER    (DIM / KCHUNK)   // 224
#define N_GROUPS 8
#define GRP_SZ   (EXPERTS / N_GROUPS)  // 32
#define TOPK_G   4
#define TOPK     8
#define RSCALE   2.5f
#define LDSROW   257               // padded row stride (floats) -> conflict-free routing reads

// ---------------- weight fp32 -> bf16 pre-convert (runs once per launch) --------------
__global__ void cvt_w_kernel(const float* __restrict__ w, __bf16* __restrict__ o, int n4) {
    int i = blockIdx.x * blockDim.x + threadIdx.x;
    if (i < n4) {
        v4f f = *(const v4f*)(w + (size_t)i * 4);
        v4bf b = __builtin_convertvector(f, v4bf);
        *(v4bf*)(o + (size_t)i * 4) = b;
    }
}

// ---------------- fused router GEMM + sigmoid + group-limited top-k -------------------
__global__ __launch_bounds__(256)
void gate_kernel(const float* __restrict__ x,
                 const __bf16* __restrict__ wbf,
                 const float* __restrict__ bias,
                 float* __restrict__ out_w,
                 int*   __restrict__ out_i)
{
    __shared__ float sc[16 * LDSROW];   // logits -> sigmoid scores, 16 tokens x 256 experts
    __shared__ float bsh[EXPERTS];

    const int tid  = threadIdx.x;
    const int lane = tid & 31;
    const int wv   = tid >> 5;          // wave 0..7 -> experts [wv*32, wv*32+32)
    const int ln16 = lane & 15;
    const int kh   = lane >> 4;         // which K-half this lane holds
    const int tb   = blockIdx.x;        // 16-token block

    const float* pa = x + (size_t)(tb * 16 + ln16) * DIM;   // A row for this lane
    const int   e0  = wv * 32 + ln16;                       // expert column, tile 0
    const __bf16* pb0 = wbf + (size_t)e0 * DIM + kh * 16;   // 16 consecutive K per lane
    const __bf16* pb1 = pb0 + (size_t)16 * DIM;             // tile 1 (+16 experts)
    const int a0 = kh * 16;         // A run 0: K = a0..a0+7
    const int a1 = kh * 16 + 8;     // A run 1: K = a1..a1+7

    v8f c0 = {};
    v8f c1 = {};

    for (int it = 0; it < NITER; ++it) {
        const int k0 = it * KCHUNK;

        // ---- A fragment: 16 fp32 -> bf16 (row ln16, runs {a0..a0+7, a1..a1+7}) ----
        v4f f0 = *(const v4f*)(pa + k0 + a0);
        v4f f1 = *(const v4f*)(pa + k0 + a0 + 4);
        v4f f2 = *(const v4f*)(pa + k0 + a1);
        v4f f3 = *(const v4f*)(pa + k0 + a1 + 4);
        v4bf g0 = __builtin_convertvector(f0, v4bf);
        v4bf g1 = __builtin_convertvector(f1, v4bf);
        v4bf g2 = __builtin_convertvector(f2, v4bf);
        v4bf g3 = __builtin_convertvector(f3, v4bf);
        v8bf h0 = __builtin_shufflevector(g0, g1, 0,1,2,3,4,5,6,7);
        v8bf h1 = __builtin_shufflevector(g2, g3, 0,1,2,3,4,5,6,7);
        v16bf a = __builtin_shufflevector(h0, h1, 0,1,2,3,4,5,6,7,8,9,10,11,12,13,14,15);

        // ---- B fragments: bf16 direct, 16 consecutive K per lane ----
        v8bf blo0 = *(const v8bf*)(pb0 + k0);
        v8bf bhi0 = *(const v8bf*)(pb0 + k0 + 8);
        v8bf blo1 = *(const v8bf*)(pb1 + k0);
        v8bf bhi1 = *(const v8bf*)(pb1 + k0 + 8);
        v16bf b0 = __builtin_shufflevector(blo0, bhi0, 0,1,2,3,4,5,6,7,8,9,10,11,12,13,14,15);
        v16bf b1 = __builtin_shufflevector(blo1, bhi1, 0,1,2,3,4,5,6,7,8,9,10,11,12,13,14,15);

        c0 = __builtin_amdgcn_wmma_f32_16x16x32_bf16(false, a, false, b0, (short)0, c0, false, false);
        c1 = __builtin_amdgcn_wmma_f32_16x16x32_bf16(false, a, false, b1, (short)0, c1, false, false);
    }

    // ---- scatter C tiles into LDS: lane holds col N=ln16, vgpr g holds row M=g+8*kh ----
    {
        const int row0 = kh * 8;
        const int col0 = wv * 32 + ln16;
        const int col1 = col0 + 16;
#pragma unroll
        for (int g = 0; g < 8; ++g) {
            sc[(row0 + g) * LDSROW + col0] = c0[g];
            sc[(row0 + g) * LDSROW + col1] = c1[g];
        }
    }
    __syncthreads();

    // ---- sigmoid in place + stage bias ----
    bsh[tid] = bias[tid];
#pragma unroll
    for (int j = 0; j < 16; ++j) {
        int li = tid * 16 + j;          // 4096 logits / 256 threads
        int m  = li >> 8;
        int e  = li & 255;
        float v = sc[m * LDSROW + e];
        sc[m * LDSROW + e] = 1.0f / (1.0f + __expf(-v));
    }
    __syncthreads();

    // ---- routing: lanes 0..15 of wave 0, one token each ----
    if (tid < 16) {
        const int m = tid;
        float* row = sc + m * LDSROW;

        // group scores: sum of top-2 of (sigmoid + bias) in each group of 32
        float gs[N_GROUPS];
#pragma unroll
        for (int g = 0; g < N_GROUPS; ++g) {
            float m1 = -1e30f, m2 = -1e30f;
            for (int j = 0; j < GRP_SZ; ++j) {
                int e = g * GRP_SZ + j;
                float s = row[e] + bsh[e];
                if (s > m1)      { m2 = m1; m1 = s; }
                else if (s > m2) { m2 = s; }
            }
            gs[g] = m1 + m2;
        }

        // top-4 groups (strict > == lowest-index tie-break, like jax top_k)
        unsigned gmask = 0;
#pragma unroll
        for (int t = 0; t < TOPK_G; ++t) {
            float best = -1e30f; int bg = 0;
#pragma unroll
            for (int g = 0; g < N_GROUPS; ++g) {
                if (!((gmask >> g) & 1u) && gs[g] > best) { best = gs[g]; bg = g; }
            }
            gmask |= 1u << bg;
        }

        // top-8 experts over masked scores; gather ORIGINAL sigmoid for weights
        float w[TOPK]; int idx[TOPK]; float wsum = 0.0f;
#pragma unroll
        for (int t = 0; t < TOPK; ++t) {
            float best = -1e30f; int be = 0;
            for (int e = 0; e < EXPERTS; ++e) {
                float scv = row[e];
                float s = ((gmask >> (e >> 5)) & 1u) ? (scv + bsh[e]) : 0.0f;
                if (s > best) { best = s; be = e; }
            }
            idx[t] = be;
            float orig = row[be];
            w[t] = orig;
            wsum += orig;
            row[be] = -1e30f;   // exclude from later passes
        }

        const float inv = RSCALE / wsum;
        const int tok = tb * 16 + m;
#pragma unroll
        for (int t = 0; t < TOPK; ++t) {
            out_w[tok * TOPK + t] = w[t] * inv;
            out_i[tok * TOPK + t] = idx[t];
        }
    }
}

extern "C" void kernel_launch(void* const* d_in, const int* in_sizes, int n_in,
                              void* d_out, int out_size, void* d_ws, size_t ws_size,
                              hipStream_t stream) {
    const float* x    = (const float*)d_in[0];   // [8192, 7168]
    const float* wgt  = (const float*)d_in[1];   // [256, 7168]
    const float* bias = (const float*)d_in[2];   // [256]

    float* out_w = (float*)d_out;                              // [8192, 8] fp32
    int*   out_i = (int*)((float*)d_out + (size_t)TOKENS * TOPK); // [8192, 8] int32

    __bf16* wbf = (__bf16*)d_ws;                 // 256*7168*2 = 3.67 MB scratch

    const int n4 = EXPERTS * DIM / 4;
    cvt_w_kernel<<<(n4 + 255) / 256, 256, 0, stream>>>(wgt, wbf, n4);
    gate_kernel<<<TOKENS / 16, 256, 0, stream>>>(x, wbf, bias, out_w, out_i);
}
